// FourierKANLayer_75350906241070
// MI455X (gfx1250) — matrix operand order repaired and checked by reading, not compiled, not run on hardware
//
#include <hip/hip_runtime.h>
#include <stdint.h>

typedef __attribute__((ext_vector_type(16))) _Float16 v16h;
typedef __attribute__((ext_vector_type(8)))  _Float16 v8h;
typedef __attribute__((ext_vector_type(8)))  float    v8f;
typedef __attribute__((ext_vector_type(4)))  uint32_t u32x4;
typedef __attribute__((ext_vector_type(8)))  uint32_t u32x8;

#define IN_DIM   256
#define OUT_DIM  256
#define GRID_SZ  32
#define NROWS    4096
#define KTILES   512                  // one K-tile = 32 K values = one (c, i)
#define NT       16                   // 16 N-tiles of 16 cols
#define FRAG_HALFS 512                // one B fragment = 32 lanes * 16 halfs = 1 KB
#define KT_HALFS (NT * FRAG_HALFS)    // 8192 halfs = 16 KB per K-tile
#define PAIR_HALFS (2 * KT_HALFS)     // TDM moves 2 K-tiles (32 KB) per descriptor
#define PAIR_BYTES (PAIR_HALFS * 2)

// ---------------------------------------------------------------------------
// Pack fouriercoeffs (f32 [2,out,in,grid]) -> f16 WMMA B fragments.
// B 32x16 per-lane layout used here (contiguous k): lane -> col n = lane%16, half = lane/16
// holds K = 16*half + e.  Intra-fragment split low/high so GEMM reads are two
// conflict-free ds_load_b128 at lane*16B stride:
//   off(lane, e) = (e<8 ? 0 : 256) + lane*8 + (e&7)   [halfs]
// K-tile kt = c*256 + i is one contiguous 16 KB chunk -> 1-D TDM tiles.
// ---------------------------------------------------------------------------
__global__ __launch_bounds__(256) void pack_b(const float* __restrict__ coeffs,
                                              _Float16* __restrict__ bpack) {
  int idx  = blockIdx.x * 256 + threadIdx.x;   // 0 .. 4194303
  int off  = idx & 511;
  int frag = idx >> 9;
  int nt   = frag & 15;
  int kt   = frag >> 4;
  int sub  = off >> 8;            // 0 -> e=0..7, 1 -> e=8..15
  int r    = off & 255;
  int lane = r >> 3;
  int e    = sub * 8 + (r & 7);
  int half = lane >> 4;
  int n    = nt * 16 + (lane & 15);
  int K    = kt * 32 + 16 * half + e;
  int c    = K >> 13;             // 0 = cos coeffs, 1 = sin coeffs
  int rem  = K & 8191;
  int i    = rem >> 5;
  int g    = rem & 31;
  bpack[idx] = (_Float16)coeffs[(((size_t)c * OUT_DIM + n) * IN_DIM + i) * GRID_SZ + g];
}

// ---------------------------------------------------------------------------
// TDM: DMA a 32 KB pair of K-tiles of packed B into LDS (1-D, 8192 dwords).
// D# per CDNA5 ISA (group0 + group1, groups 2/3 NULL).
// ---------------------------------------------------------------------------
__device__ __forceinline__ void tdm_load_pair(uint32_t lds_off,
                                              const _Float16* gsrc) {
  uint64_t ga = (uint64_t)(uintptr_t)gsrc;
  u32x4 g0;
  g0.x = 0x1u;                                 // count=1 (valid descriptor)
  g0.y = lds_off;                              // LDS byte address
  g0.z = (uint32_t)ga;                         // global_addr[31:0]
  g0.w = (uint32_t)(ga >> 32) | (2u << 30);    // global_addr[56:32] | type=2
  u32x8 g1;
  g1[0] = 2u << 16;          // workgroup_mask=0, data_size=2 (4-byte elements)
  g1[1] = 8192u << 16;       // tensor_dim0[15:0] = 8192 dwords
  g1[2] = 1u << 16;          // tensor_dim0[31:16]=0, tensor_dim1[15:0]=1
  g1[3] = 8192u << 16;       // tensor_dim1[31:16]=0, tile_dim0 = 8192 dwords
  g1[4] = 1u;                // tile_dim1 = 1, tile_dim2 = 0
  g1[5] = 8192u;             // tensor_dim0_stride[31:0]
  g1[6] = 0u;
  g1[7] = 0u;
  asm volatile("tensor_load_to_lds %0, %1" : : "s"(g0), "s"(g1) : "memory");
}

// ---------------------------------------------------------------------------
// GEMM: out[m][n] = sum_{i,g} cos(x[m][i](g+1)) C0[n][i][g]
//                 + sum_{i,g} sin(x[m][i](g+1)) C1[n][i][g] + bias[n]
// 256 threads = 8 waves. Wave tile = 2 M-tiles x 4 N-tiles (32 rows x 64 cols):
//   mgrp = wave & 1  -> rows  blockIdx*64 + mgrp*32 + {0,16}
//   ngrp = wave >> 1 -> cols  ngrp*64 .. +63
// Per K-tile each wave loads only 4 B fragments (4 KB LDS) feeding 8 WMMAs,
// and builds 2 A fragments from on-the-fly v_cos/v_sin.
// TDM double-buffers 32 KB (2 K-tiles) one stage ahead; barrier every 2 tiles.
// ---------------------------------------------------------------------------
__global__ __launch_bounds__(256) void fourier_gemm(
    const float* __restrict__ x, const _Float16* __restrict__ bpack,
    const float* __restrict__ bias, float* __restrict__ out) {
  __shared__ __align__(16) _Float16 lds_b[2 * PAIR_HALFS];   // 64 KB double buffer

  const int lane  = threadIdx.x & 31;
  const int wave  = threadIdx.x >> 5;
  const int mgrp  = wave & 1;
  const int ngrp  = wave >> 1;
  const int half  = lane >> 4;
  const int l16   = lane & 15;
  const int m0    = blockIdx.x * 64 + mgrp * 32;

  const uint32_t lds_base = (uint32_t)(uintptr_t)&lds_b[0];

  v8f acc[8];                       // [mt*4 + t]
#pragma unroll
  for (int t = 0; t < 8; ++t)
    acc[t] = (v8f){0.f, 0.f, 0.f, 0.f, 0.f, 0.f, 0.f, 0.f};

  const float* xp0 = x + (size_t)(m0 + l16) * IN_DIM;        // rows m0..m0+15
  const float* xp1 = x + (size_t)(m0 + 16 + l16) * IN_DIM;   // rows m0+16..m0+31

  // prologue: DMA K-tile pair 0 into buffer 0
  if (wave == 0) {
    tdm_load_pair(lds_base, bpack);
    __builtin_amdgcn_s_wait_tensorcnt(0);
  }
  __syncthreads();

  // A fragment builder: 16-bit A 16x32 per-lane, k = 16*half + e (same order as packed B)
  auto build_a = [&](float s, bool is_sin) __attribute__((always_inline)) -> v16h {
    v16h a;
#pragma unroll
    for (int e = 0; e < 16; ++e) {
      int v   = e >> 1, pos = e & 1;
      int gk  = 16 * half + 2 * v + pos;
      float ph = s * (float)(gk + 1);
      float tv = is_sin ? __sinf(ph) : __cosf(ph);
      a[e] = (_Float16)tv;
    }
    return a;
  };

  // one K-tile of compute out of LDS buffer p, sub-slot sub (0/1)
  auto kbody = [&](int kt, int p, int sub, bool is_sin) __attribute__((always_inline)) {
    const int i = kt & (IN_DIM - 1);
    v16h a0 = build_a(xp0[i], is_sin);
    v16h a1 = build_a(xp1[i], is_sin);

    const _Float16* lb =
        &lds_b[p * PAIR_HALFS + sub * KT_HALFS + (ngrp * 4) * FRAG_HALFS + lane * 8];
#pragma unroll
    for (int t = 0; t < 4; ++t) {
      v8h lo = *(const v8h*)(lb + t * FRAG_HALFS);
      v8h hi = *(const v8h*)(lb + t * FRAG_HALFS + 256);
      v16h b = __builtin_shufflevector(lo, hi, 0, 1, 2, 3, 4, 5, 6, 7,
                                       8, 9, 10, 11, 12, 13, 14, 15);
      acc[t]     = __builtin_amdgcn_wmma_f32_16x16x32_f16(
          false, a0, false, b, (short)0, acc[t], false, false);
      acc[4 + t] = __builtin_amdgcn_wmma_f32_16x16x32_f16(
          false, a1, false, b, (short)0, acc[4 + t], false, false);
    }
  };

  auto pair_body = [&](int ktp, bool is_sin) __attribute__((always_inline)) {
    const int p = ktp & 1;
    if (wave == 0 && ktp + 1 < KTILES / 2)
      tdm_load_pair(lds_base + (uint32_t)((p ^ 1) * PAIR_BYTES),
                    bpack + (size_t)(ktp + 1) * PAIR_HALFS);
    kbody(2 * ktp,     p, 0, is_sin);
    kbody(2 * ktp + 1, p, 1, is_sin);
    if (wave == 0) __builtin_amdgcn_s_wait_tensorcnt(0);
    __syncthreads();
  };

  for (int ktp = 0;   ktp < 128; ++ktp) pair_body(ktp, false);  // cos half
  for (int ktp = 128; ktp < 256; ++ktp) pair_body(ktp, true);   // sin half

  // epilogue: C/D 16x16 f32 layout -> lane = col n, VGPR r -> row r + 8*half
#pragma unroll
  for (int mt = 0; mt < 2; ++mt) {
#pragma unroll
    for (int t = 0; t < 4; ++t) {
      int n    = ngrp * 64 + t * 16 + l16;
      float bv = bias[n];
#pragma unroll
      for (int r = 0; r < 8; ++r) {
        int m = m0 + mt * 16 + r + 8 * half;
        out[(size_t)m * OUT_DIM + n] = acc[mt * 4 + t][r] + bv;
      }
    }
  }
}

extern "C" void kernel_launch(void* const* d_in, const int* in_sizes, int n_in,
                              void* d_out, int out_size, void* d_ws, size_t ws_size,
                              hipStream_t stream) {
  (void)in_sizes; (void)n_in; (void)out_size; (void)ws_size;
  const float* x      = (const float*)d_in[0];   // [4096, 256] f32
  const float* coeffs = (const float*)d_in[1];   // [2, 256, 256, 32] f32
  const float* bias   = (const float*)d_in[2];   // [1, 256] f32
  float*       out    = (float*)d_out;           // [4096, 256] f32
  _Float16*    bpack  = (_Float16*)d_ws;         // 8 MiB packed f16 B fragments

  const int pack_elems = KTILES * NT * FRAG_HALFS;    // 4,194,304
  pack_b<<<pack_elems / 256, 256, 0, stream>>>(coeffs, bpack);

  fourier_gemm<<<NROWS / 64, 256, 0, stream>>>(x, bpack, bias, out);
}
